// LlamaAttentionExperimental_20469814133367
// MI455X (gfx1250) — matrix operand order repaired and checked by reading, not compiled, not run on hardware
//
#include <hip/hip_runtime.h>
#include <stdint.h>

typedef unsigned short u16;
typedef unsigned int   u32;
typedef unsigned long long u64;
typedef __attribute__((ext_vector_type(4)))  u32    v4u;
typedef __attribute__((ext_vector_type(4)))  int    v4i;
typedef __attribute__((ext_vector_type(8)))  int    v8i;
typedef __attribute__((ext_vector_type(16))) __bf16 v16bf;
typedef __attribute__((ext_vector_type(8)))  float  v8f;

union Frag { v16bf v; v4u q[2]; };

__device__ __forceinline__ u16 f2bf(float f) {
  union { float f; u32 u; } c; c.f = f;
  u32 u = c.u;
  u32 r = u + 0x7FFFu + ((u >> 16) & 1u);   // round-to-nearest-even
  return (u16)(r >> 16);
}
__device__ __forceinline__ float bf2f(u16 h) {
  union { u32 u; float f; } c; c.u = ((u32)h) << 16;
  return c.f;
}
__device__ __forceinline__ float rmax16(float v) {
#pragma unroll
  for (int m = 1; m < 16; m <<= 1) v = fmaxf(v, __shfl_xor(v, m, 16));
  return v;
}
__device__ __forceinline__ float rsum16(float v) {
#pragma unroll
  for (int m = 1; m < 16; m <<= 1) v += __shfl_xor(v, m, 16);
  return v;
}

// ---- Tensor Data Mover: issue a 2-D tile load (rows x 64 bf16) into LDS ----
// D# per CDNA5 ISA ch.8: group0 = {count/type/addr}, group1 = {dims/strides/pad}.
// Pads every 32 DWORDs (one 64-elem row) by 4 DWORDs -> LDS row stride 72 elems
// (144 B: 16B-aligned for ds_load_b128, bank-conflict-free across 16 lanes).
__device__ __forceinline__ void tdm_load_tile64(const u16* gsrc, u32 lds_off,
                                                u32 rows, u32 row_stride_elems) {
  u64 ga = (u64)(uintptr_t)gsrc;
  v4u g0;
  g0[0] = 1u;                                        // count=1, no gather
  g0[1] = lds_off;                                   // lds_addr (bytes)
  g0[2] = (u32)ga;                                   // global_addr[31:0]
  g0[3] = (u32)((ga >> 32) & 0x1FFFFFFu) | (2u << 30); // addr[56:32] | type=2
  v8i g1;
  g1[0] = (int)((1u << 16)        // data_size = 2 bytes
              | (1u << 20)        // pad_enable
              | (4u << 22)        // pad_interval: 32 DWORDs
              | (3u << 25));      // pad_amount:   4 DWORDs
  g1[1] = (int)(64u << 16);                 // tensor_dim0[15:0] = 64
  g1[2] = (int)((rows & 0xFFFFu) << 16);    // tensor_dim0 hi=0 | tensor_dim1 lo
  g1[3] = (int)(64u << 16);                 // tensor_dim1 hi=0 | tile_dim0 = 64
  g1[4] = (int)rows;                        // tile_dim1 = rows, tile_dim2 = 0
  g1[5] = (int)row_stride_elems;            // tensor_dim0_stride[31:0]
  g1[6] = 0;
  g1[7] = 0;
  v4i z4 = {0, 0, 0, 0};
  v8i z8 = {0, 0, 0, 0, 0, 0, 0, 0};
  __builtin_amdgcn_tensor_load_to_lds(g0, g1, z4, z4, z8, 0);
}

// ---------------------------------------------------------------- f32 -> bf16
__global__ void __launch_bounds__(256) cvt_f32_bf16(const float* __restrict__ in,
                                                    u16* __restrict__ out, int n4) {
  int i = blockIdx.x * 256 + threadIdx.x;
  if (i >= n4) return;
  float4 v = ((const float4*)in)[i];
  ((u32*)out)[i * 2 + 0] = (u32)f2bf(v.x) | ((u32)f2bf(v.y) << 16);
  ((u32*)out)[i * 2 + 1] = (u32)f2bf(v.z) | ((u32)f2bf(v.w) << 16);
}

// --------------------------------------------- C[M,N] = A[M,K] * B[N,K]^T
// Block tile 128x128, k-tile 64, TDM-staged LDS, double-buffered.
// 8 waves: wave tile 32x64 (2 A-frags x 4 B-frags -> 8 WMMA per k-step).
__global__ void __launch_bounds__(256) gemm_bf16nt_tdm(const u16* __restrict__ A,
                                                       const u16* __restrict__ Bw,
                                                       void* __restrict__ Cout,
                                                       int M, int N, int K, int store_bf16) {
  __shared__ u16 As[2][128 * 72];
  __shared__ u16 Bs[2][128 * 72];

  const int tid = threadIdx.x;
  const int wv = tid >> 5, lane = tid & 31;
  const int half = lane >> 4, mn = lane & 15;
  const int nb = N >> 7;
  const int by = blockIdx.x / nb, bx = blockIdx.x - by * nb;
  const int rowBase = by * 128, colBase = bx * 128;
  const int mw = wv >> 1, nw = wv & 1;                 // 4x2 wave grid

  const u32 offA[2] = { (u32)(uintptr_t)&As[0][0], (u32)(uintptr_t)&As[1][0] };
  const u32 offB[2] = { (u32)(uintptr_t)&Bs[0][0], (u32)(uintptr_t)&Bs[1][0] };

  const u16* aG = A  + (size_t)rowBase * K;
  const u16* bG = Bw + (size_t)colBase * K;

  v8f acc[2][4] = {};
  const int nk = K >> 6;

  if (wv == 0) {                                        // TDM ignores EXEC; wave-uniform
    tdm_load_tile64(aG, offA[0], 128, (u32)K);
    tdm_load_tile64(bG, offB[0], 128, (u32)K);
    __builtin_amdgcn_s_wait_tensorcnt(0);
  }
  __syncthreads();

  for (int kt = 0; kt < nk; kt++) {
    const int buf = kt & 1;
    if (wv == 0 && kt + 1 < nk) {                       // prefetch next k-tile via TDM
      tdm_load_tile64(aG + (size_t)(kt + 1) * 64, offA[buf ^ 1], 128, (u32)K);
      tdm_load_tile64(bG + (size_t)(kt + 1) * 64, offB[buf ^ 1], 128, (u32)K);
    }
    const u16* as = As[buf];
    const u16* bs = Bs[buf];
#pragma unroll
    for (int ks = 0; ks < 2; ks++) {
      Frag af[2];
#pragma unroll
      for (int ms = 0; ms < 2; ms++) {
        const u16* ap = as + (mw * 32 + ms * 16 + mn) * 72 + ks * 32 + half * 8;
        af[ms].q[0] = *(const v4u*)ap;
        af[ms].q[1] = *(const v4u*)(ap + 16);
      }
#pragma unroll
      for (int t = 0; t < 4; t++) {
        Frag bf;
        const u16* bp = bs + (nw * 64 + t * 16 + mn) * 72 + ks * 32 + half * 8;
        bf.q[0] = *(const v4u*)bp;
        bf.q[1] = *(const v4u*)(bp + 16);
#pragma unroll
        for (int ms = 0; ms < 2; ms++)
          acc[ms][t] = __builtin_amdgcn_wmma_f32_16x16x32_bf16(false, af[ms].v, false, bf.v,
                                                               (short)0, acc[ms][t], false, false);
      }
    }
    __syncthreads();                                    // everyone done with buf
    if (wv == 0 && kt + 1 < nk) __builtin_amdgcn_s_wait_tensorcnt(0);
    __syncthreads();                                    // next buffer ready for all
  }

#pragma unroll
  for (int ms = 0; ms < 2; ms++) {
#pragma unroll
    for (int t = 0; t < 4; t++) {
#pragma unroll
      for (int r = 0; r < 8; r++) {
        size_t ro = (size_t)(rowBase + mw * 32 + ms * 16 + r + half * 8);
        size_t co = (size_t)(colBase + nw * 64 + t * 16 + mn);
        if (store_bf16) ((u16*)Cout)[ro * N + co] = f2bf(acc[ms][t][r]);
        else            ((float*)Cout)[ro * N + co] = acc[ms][t][r];
      }
    }
  }
}

// ----------------------------------------------------------------- RoPE (bf16)
__global__ void __launch_bounds__(256) rope_bf16(u16* __restrict__ X,
                                                 const float* __restrict__ cosT,
                                                 const float* __restrict__ sinT,
                                                 int width, int S, int rows) {
  int ppr = width >> 1;
  long long idx = (long long)blockIdx.x * 256 + threadIdx.x;
  if (idx >= (long long)rows * ppr) return;
  int row = (int)(idx / ppr);
  int p   = (int)(idx - (long long)row * ppr);
  int h = p >> 6, d = p & 63;
  int s = row & (S - 1);
  size_t base = (size_t)row * width + h * 128;
  float x1 = bf2f(X[base + d]),      x2 = bf2f(X[base + d + 64]);
  float c1 = cosT[(size_t)s * 128 + d],      s1 = sinT[(size_t)s * 128 + d];
  float c2 = cosT[(size_t)s * 128 + d + 64], s2 = sinT[(size_t)s * 128 + d + 64];
  X[base + d]      = f2bf(x1 * c1 - x2 * s1);
  X[base + d + 64] = f2bf(x2 * c2 + x1 * s2);
}

// -------------------------------------------------------- flash attention (GQA)
__global__ void __launch_bounds__(256) flash_attn(const u16* __restrict__ Qb,
                                                  const u16* __restrict__ Kb,
                                                  const u16* __restrict__ Vb,
                                                  u16* __restrict__ Ab) {
  const int S = 2048, QD = 4096, KVD = 1024;
  const float scale = 0.08838834764831845f;         // 1/sqrt(128)
  __shared__ u16 Ks[64 * 128];                      // [key][d]
  __shared__ u16 Vt[128 * 72];                      // [d][key], padded stride
  __shared__ u16 Ps[8 * 16 * 72];                   // per-wave P scratch

  const int tid = threadIdx.x;
  const int wv = tid >> 5, lane = tid & 31;
  const int half = lane >> 4, mn = lane & 15;
  const int qblk = blockIdx.x, h = blockIdx.y, b = blockIdx.z;
  const int kvh = h >> 2;                           // GROUPS = 4
  const int q0 = qblk * 128;

  Frag qa[4];                                       // Q tile 16x128 in A layout
  {
    const u16* qrow = Qb + (size_t)(b * S + q0 + wv * 16 + mn) * QD + h * 128;
#pragma unroll
    for (int kd = 0; kd < 4; kd++) {
      const u16* qp = qrow + kd * 32 + half * 8;
      qa[kd].q[0] = *(const v4u*)qp;
      qa[kd].q[1] = *(const v4u*)(qp + 16);
    }
  }

  v8f o_acc[8] = {};
  float m_r[8], l_r[8];
#pragma unroll
  for (int r = 0; r < 8; r++) { m_r[r] = -1e30f; l_r[r] = 0.f; }

  const int kmax = q0 + 127;
  for (int k0 = 0; k0 <= kmax; k0 += 64) {
#pragma unroll
    for (int j = 0; j < 4; j++) {
      int c = j * 256 + tid;
      int e = c * 8;
      int kk = e >> 7, dd = e & 127;
      size_t g = (size_t)(b * S + k0 + kk) * KVD + kvh * 128 + dd;
      *(v4u*)&Ks[kk * 128 + dd] = *(const v4u*)(Kb + g);
      v4u vv = *(const v4u*)(Vb + g);
      const u16* pv = (const u16*)&vv;
#pragma unroll
      for (int t = 0; t < 8; t++) Vt[(dd + t) * 72 + kk] = pv[t];
    }
    __syncthreads();

    v8f s_acc[4] = {};
#pragma unroll
    for (int kd = 0; kd < 4; kd++) {
#pragma unroll
      for (int nt = 0; nt < 4; nt++) {
        Frag kb;
        const u16* kp = &Ks[(nt * 16 + mn) * 128 + kd * 32 + half * 8];
        kb.q[0] = *(const v4u*)kp;
        kb.q[1] = *(const v4u*)(kp + 16);
        s_acc[nt] = __builtin_amdgcn_wmma_f32_16x16x32_bf16(false, qa[kd].v, false, kb.v,
                                                            (short)0, s_acc[nt], false, false);
      }
    }

#pragma unroll
    for (int nt = 0; nt < 4; nt++) {
      int col = k0 + nt * 16 + mn;
#pragma unroll
      for (int r = 0; r < 8; r++) {
        int row = q0 + wv * 16 + r + half * 8;
        float sv = s_acc[nt][r] * scale;
        s_acc[nt][r] = (col <= row) ? sv : -1e30f;
      }
    }
    u16* ps = &Ps[wv * 16 * 72];
#pragma unroll
    for (int r = 0; r < 8; r++) {
      float vmax = fmaxf(fmaxf(s_acc[0][r], s_acc[1][r]), fmaxf(s_acc[2][r], s_acc[3][r]));
      vmax = rmax16(vmax);
      float mnew  = fmaxf(m_r[r], vmax);
      float alpha = __expf(m_r[r] - mnew);
      float psum = 0.f;
#pragma unroll
      for (int nt = 0; nt < 4; nt++) {
        float p = __expf(s_acc[nt][r] - mnew);
        psum += p;
        ps[(r + half * 8) * 72 + nt * 16 + mn] = f2bf(p);
      }
      psum = rsum16(psum);
      l_r[r] = l_r[r] * alpha + psum;
      m_r[r] = mnew;
#pragma unroll
      for (int dt = 0; dt < 8; dt++) o_acc[dt][r] *= alpha;
    }

#pragma unroll
    for (int ks = 0; ks < 2; ks++) {
      Frag pa;
      const u16* pp = ps + mn * 72 + ks * 32 + half * 8;
      pa.q[0] = *(const v4u*)pp;
      pa.q[1] = *(const v4u*)(pp + 16);
#pragma unroll
      for (int dt = 0; dt < 8; dt++) {
        Frag vb;
        const u16* vp = &Vt[(dt * 16 + mn) * 72 + ks * 32 + half * 8];
        vb.q[0] = *(const v4u*)vp;
        vb.q[1] = *(const v4u*)(vp + 16);
        o_acc[dt] = __builtin_amdgcn_wmma_f32_16x16x32_bf16(false, pa.v, false, vb.v,
                                                            (short)0, o_acc[dt], false, false);
      }
    }
    __syncthreads();
  }

  size_t rowO = (size_t)(b * S + q0 + wv * 16);
#pragma unroll
  for (int dt = 0; dt < 8; dt++) {
#pragma unroll
    for (int r = 0; r < 8; r++) {
      float ov = o_acc[dt][r] / l_r[r];
      Ab[(rowO + r + half * 8) * QD + h * 128 + dt * 16 + mn] = f2bf(ov);
    }
  }
}

// ---------------------------------------------------------------------- launch
extern "C" void kernel_launch(void* const* d_in, const int* in_sizes, int n_in,
                              void* d_out, int out_size, void* d_ws, size_t ws_size,
                              hipStream_t stream) {
  const float* hs   = (const float*)d_in[0];
  const float* Wq   = (const float*)d_in[1];
  const float* Wk   = (const float*)d_in[2];
  const float* Wv   = (const float*)d_in[3];
  const float* Wo   = (const float*)d_in[4];
  const float* cosT = (const float*)d_in[5];
  const float* sinT = (const float*)d_in[6];
  float* out = (float*)d_out;

  const int BS = 4096, H = 4096, QD = 4096, KVD = 1024, S = 2048;

  u16* ws = (u16*)d_ws;
  size_t o = 0;
  u16* Xb  = ws + o; o += (size_t)BS * H;
  u16* Wqb = ws + o; o += (size_t)QD * H;
  u16* Wkb = ws + o; o += (size_t)KVD * H;
  u16* Wvb = ws + o; o += (size_t)KVD * H;
  u16* Wob = ws + o; o += (size_t)H * QD;
  u16* Qb  = ws + o; o += (size_t)BS * QD;
  u16* Kb  = ws + o; o += (size_t)BS * KVD;
  u16* Vb  = ws + o; o += (size_t)BS * KVD;
  u16* Ab  = ws + o;

  auto cv = [&](const float* src, u16* dst, size_t n) {
    int n4 = (int)(n / 4);
    cvt_f32_bf16<<<(n4 + 255) / 256, 256, 0, stream>>>(src, dst, n4);
  };
  cv(hs, Xb, (size_t)BS * H);
  cv(Wq, Wqb, (size_t)QD * H);
  cv(Wk, Wkb, (size_t)KVD * H);
  cv(Wv, Wvb, (size_t)KVD * H);
  cv(Wo, Wob, (size_t)H * QD);

  gemm_bf16nt_tdm<<<(BS / 128) * (QD / 128), 256, 0, stream>>>(Xb, Wqb, Qb, BS, QD, H, 1);
  gemm_bf16nt_tdm<<<(BS / 128) * (KVD / 128), 256, 0, stream>>>(Xb, Wkb, Kb, BS, KVD, H, 1);
  gemm_bf16nt_tdm<<<(BS / 128) * (KVD / 128), 256, 0, stream>>>(Xb, Wvb, Vb, BS, KVD, H, 1);

  {
    long long tq = (long long)BS * (QD / 2);
    rope_bf16<<<(int)((tq + 255) / 256), 256, 0, stream>>>(Qb, cosT, sinT, QD, S, BS);
    long long tk = (long long)BS * (KVD / 2);
    rope_bf16<<<(int)((tk + 255) / 256), 256, 0, stream>>>(Kb, cosT, sinT, KVD, S, BS);
  }

  dim3 fg(S / 128, 32, 2);
  flash_attn<<<fg, 256, 0, stream>>>(Qb, Kb, Vb, Ab);

  gemm_bf16nt_tdm<<<(BS / 128) * (QD / 128), 256, 0, stream>>>(Ab, Wob, out, BS, QD, H, 0);
}
